// StockNet_17832704213432
// MI455X (gfx1250) — matrix operand order repaired and checked by reading, not compiled, not run on hardware
//
#include <hip/hip_runtime.h>
#include <hip/hip_bf16.h>
#include <math.h>

// ---------------------------------------------------------------------------
// Problem constants (from the reference)
// ---------------------------------------------------------------------------
#define BB 8
#define SS 500
#define TT 30
#define DD 5
#define KK 10
#define HH 64
#define TW 512
#define EE 8000
#define NBS  (BB * SS)          // 4000
#define NTW  (BB * SS * DD)     // 20000
#define G3H  (3 * HH)           // 192

typedef __attribute__((ext_vector_type(16))) _Float16 v16h;
typedef __attribute__((ext_vector_type(8)))  float    v8f;

__device__ __forceinline__ float sigmoidf_(float x) {
  return 1.0f / (1.0f + expf(-x));
}

// ---------------------------------------------------------------------------
// Pack B[K,N] (f32) into WMMA-fragment-major f16:
//   chunk c=k/32, tile t=n/16, lane = ((k%32)/8%2)*16 + n%16, e = k%8 + (k%32)/16*8
//   Bp[ ((c*(N/16)+t)*32 + lane)*16 + e ]
// One v16h (32B) load in the GEMM then yields a lane's whole B fragment.
// ---------------------------------------------------------------------------
__global__ void packB_kernel(const float* __restrict__ B,
                             _Float16* __restrict__ Bp,
                             int K, int N) {
  const int idx = blockIdx.x * blockDim.x + threadIdx.x;
  if (idx >= K * N) return;
  const int n = idx % N;
  const int k = idx / N;
  const int c = k >> 5, kl = k & 31;
  const int half = (kl >> 3) & 1, e = (kl & 7) + ((kl >> 4) << 3);
  const int t = n >> 4;
  const int lane = half * 16 + (n & 15);
  Bp[(((size_t)c * (N >> 4) + t) * 32 + lane) * 16 + e] = (_Float16)B[idx];
}

// ---------------------------------------------------------------------------
// Aligned GEMM: C[M,N] = act( A[M,K] @ Bp + bias[N] )
// REQUIRES: M % 16 == 0, K % 32 == 0, N % 64 == 0 (call sites guarantee this).
// One wave computes a 16x64 C strip: one A fragment feeds 4 v_wmma per chunk.
// A: f32 row-major (converted in-register); Bp: pre-packed f16 fragments.
// act: 0=none, 1=tanh, 2=relu
// ---------------------------------------------------------------------------
__global__ void gemm_wmma4_kernel(const float* __restrict__ A,
                                  const _Float16* __restrict__ Bp,
                                  const float* __restrict__ bias,
                                  float* __restrict__ C,
                                  int M, int K, int N, int act) {
  const int lane = threadIdx.x & 31;       // wave32
  const int m0 = blockIdx.y * 16;
  const int n0 = blockIdx.x * 64;
  const int tileBase = blockIdx.x * 4;     // first of 4 n-tiles
  const int ntiles = N >> 4;
  const int half = (lane < 16) ? 0 : 1;

  // A fragment (16x32 f16): half 0 -> K {0..7,16..23}+kc, half 1 -> +8
  const int aRow = m0 + (lane & 15);
  const float* Arow = A + (size_t)aRow * K + (half ? 8 : 0);
  // Packed B: lane's fragment for (chunk c, tile t) is 16 contiguous halfs.
  const _Float16* Bl = Bp + ((size_t)tileBase * 32 + lane) * 16;
  const size_t chunkStride = (size_t)ntiles * 32 * 16;   // halfs per k-chunk

  v8f acc[4] = {{}, {}, {}, {}};
  for (int kc = 0; kc < K; kc += 32) {
    __builtin_prefetch(Arow + kc + 32, 0, 1);   // speculative: dropped if OOB
    const float4 a0 = *(const float4*)(Arow + kc);
    const float4 a1 = *(const float4*)(Arow + kc + 4);
    const float4 a2 = *(const float4*)(Arow + kc + 16);
    const float4 a3 = *(const float4*)(Arow + kc + 20);
    v16h a;
    a[0]  = (_Float16)a0.x; a[1]  = (_Float16)a0.y;
    a[2]  = (_Float16)a0.z; a[3]  = (_Float16)a0.w;
    a[4]  = (_Float16)a1.x; a[5]  = (_Float16)a1.y;
    a[6]  = (_Float16)a1.z; a[7]  = (_Float16)a1.w;
    a[8]  = (_Float16)a2.x; a[9]  = (_Float16)a2.y;
    a[10] = (_Float16)a2.z; a[11] = (_Float16)a2.w;
    a[12] = (_Float16)a3.x; a[13] = (_Float16)a3.y;
    a[14] = (_Float16)a3.z; a[15] = (_Float16)a3.w;

    const _Float16* Bc = Bl + (size_t)(kc >> 5) * chunkStride;
#pragma unroll
    for (int nt = 0; nt < 4; ++nt) {
      const v16h b = *(const v16h*)(Bc + nt * (32 * 16));  // 2x b128
      acc[nt] = __builtin_amdgcn_wmma_f32_16x16x32_f16(
          /*neg_a=*/false, a, /*neg_b=*/false, b,
          /*c_mod=*/(short)0, acc[nt], /*reuse_a=*/false, /*reuse_b=*/false);
    }
  }

  // C/D layout: VGPR r -> row m0 + r + half*8, col = n0 + (lane&15) + nt*16
  const int mBase = m0 + half * 8;
  const int col = n0 + (lane & 15);
#pragma unroll
  for (int nt = 0; nt < 4; ++nt) {
    const int n = col + nt * 16;
    const float bv = bias ? bias[n] : 0.0f;
#pragma unroll
    for (int r = 0; r < 8; ++r) {
      float v = acc[nt][r] + bv;
      if (act == 1)      v = tanhf(v);
      else if (act == 2) v = fmaxf(v, 0.0f);
      C[(size_t)(mBase + r) * N + n] = v;
    }
  }
}

// ---------------------------------------------------------------------------
// Zero-pad the price branch (K=3 -> K=32) so the GEMM stays guard-free.
// ---------------------------------------------------------------------------
__global__ void pad_priceA_kernel(const float* __restrict__ price,
                                  float* __restrict__ Ap) {
  const int idx = blockIdx.x * blockDim.x + threadIdx.x;   // 120000*32
  if (idx >= NBS * TT * 32) return;
  const int c = idx & 31;
  const int row = idx >> 5;
  Ap[idx] = (c < 3) ? price[row * 3 + c] : 0.0f;
}

__global__ void pad_priceB_kernel(const float* __restrict__ Wi,   // [3,192]
                                  float* __restrict__ Bp) {       // [32,192]
  const int idx = blockIdx.x * blockDim.x + threadIdx.x;
  if (idx >= 32 * G3H) return;
  const int k = idx / G3H;
  Bp[idx] = (k < 3) ? Wi[idx] : 0.0f;   // same row-major layout for k<3
}

// ---------------------------------------------------------------------------
// GRU scan over T steps, emitting sum_t(full_t) (all the reference consumes).
// 4 sequences per 256-thread block; Wh (64x192) staged in LDS.
// lengths==nullptr -> dense; else freeze h past length, zero padded steps.
// ---------------------------------------------------------------------------
__global__ void gru_scan_kernel(const float* __restrict__ xg,   // [Nseq, T, 192]
                                const float* __restrict__ Wh,   // [64, 192]
                                const float* __restrict__ bh,   // [192]
                                const int*   __restrict__ lengths, // [Nseq] or null
                                float* __restrict__ hsum,       // [Nseq, 64]
                                int Nseq, int T) {
  __shared__ float Wh_s[HH * G3H];     // 48 KB
  __shared__ float h_s[4][HH];

  const int tid = threadIdx.x;         // 0..255
  for (int i = tid; i < HH * G3H; i += 256) Wh_s[i] = Wh[i];

  const int sl = tid >> 6;             // 0..3
  const int j  = tid & 63;             // hidden unit
  const int seq = blockIdx.x * 4 + sl; // Nseq divisible by 4 at all call sites
  h_s[sl][j] = 0.0f;
  __syncthreads();

  const float bhr = bh[j], bhz = bh[HH + j], bhn = bh[2 * HH + j];
  const int len = lengths ? lengths[seq] : T;
  const float* xgs = xg + (size_t)seq * T * G3H;

  float acc = 0.0f;
  for (int t = 0; t < T; ++t) {
    float hr = bhr, hz = bhz, hn = bhn;
#pragma unroll 16
    for (int i = 0; i < HH; ++i) {
      const float hv = h_s[sl][i];
      hr = fmaf(hv, Wh_s[i * G3H + j], hr);
      hz = fmaf(hv, Wh_s[i * G3H + HH + j], hz);
      hn = fmaf(hv, Wh_s[i * G3H + 2 * HH + j], hn);
    }
    const float xr = xgs[t * G3H + j];
    const float xz = xgs[t * G3H + HH + j];
    const float xn = xgs[t * G3H + 2 * HH + j];
    const float r = sigmoidf_(xr + hr);
    const float z = sigmoidf_(xz + hz);
    const float n = tanhf(xn + r * hn);
    const float hold = h_s[sl][j];
    float hnew = (1.0f - z) * n + z * hold;
    if (lengths) {
      const bool valid = (t < len);
      hnew = valid ? hnew : hold;
      acc += valid ? hnew : 0.0f;
    } else {
      acc += hnew;
    }
    __syncthreads();
    h_s[sl][j] = hnew;
    __syncthreads();
  }
  hsum[(size_t)seq * HH + j] = acc;
}

// ---------------------------------------------------------------------------
// bil_A (k,i,j) -> Bt (i, k*64+j) so the bilinear becomes text @ Bt.
// ---------------------------------------------------------------------------
__global__ void bilA_transpose_kernel(const float* __restrict__ A,
                                      float* __restrict__ Bt) {
  const int idx = blockIdx.x * blockDim.x + threadIdx.x;   // 64*64*64
  if (idx >= HH * HH * HH) return;
  const int j = idx & 63;
  const int i = (idx >> 6) & 63;
  const int k = idx >> 12;
  Bt[(size_t)i * (HH * HH) + k * HH + j] = A[(size_t)k * (HH * HH) + i * HH + j];
}

// comb[n,k] = tanh( sum_j tmp[n, k*64+j] * x[n,j] + bil_b[k] )
__global__ void bilin_combine_kernel(const float* __restrict__ tmp,  // [4000,4096]
                                     const float* __restrict__ x,    // [4000,64]
                                     const float* __restrict__ bb,   // [64]
                                     float* __restrict__ comb) {     // [4000,64]
  __shared__ float xs[HH];
  const int n = blockIdx.x, k = threadIdx.x;
  xs[k] = x[(size_t)n * HH + k];
  __syncthreads();
  float s = bb[k];
  const float* tp = tmp + (size_t)n * (HH * HH) + k * HH;
#pragma unroll 16
  for (int j = 0; j < HH; ++j) s = fmaf(tp[j], xs[j], s);
  comb[(size_t)n * HH + k] = tanhf(s);
}

// ---------------------------------------------------------------------------
// GNN pieces
// ---------------------------------------------------------------------------
__global__ void fill0_kernel(float* __restrict__ p, int n) {
  const int i = blockIdx.x * blockDim.x + threadIdx.x;
  if (i < n) p[i] = 0.0f;
}

__global__ void deg_kernel(const int* __restrict__ ei, float* __restrict__ deg) {
  const int e = blockIdx.x * blockDim.x + threadIdx.x;
  if (e < EE) atomicAdd(&deg[ei[EE + e]], 1.0f);
}

// agg[b, dst[e], h] += comb[b*S + src[e], h]
__global__ void agg_kernel(const int* __restrict__ ei,
                           const float* __restrict__ comb,
                           float* __restrict__ agg) {
  const int e = blockIdx.x, b = blockIdx.y, h = threadIdx.x;
  const int s = ei[e], d = ei[EE + e];
  atomicAdd(&agg[((size_t)b * SS + d) * HH + h],
            comb[((size_t)b * SS + s) * HH + h]);
}

__global__ void agg_norm_kernel(const float* __restrict__ agg,
                                const float* __restrict__ deg,
                                float* __restrict__ aggn) {
  const int n = blockIdx.x, h = threadIdx.x;      // n in [0,4000)
  const int s = n % SS;
  aggn[(size_t)n * HH + h] = agg[(size_t)n * HH + h] / fmaxf(deg[s], 1.0f);
}

__global__ void concat_kernel(const float* __restrict__ ft,
                              const float* __restrict__ g,
                              float* __restrict__ hcat) {
  const int idx = blockIdx.x * blockDim.x + threadIdx.x;  // 4000*128
  if (idx >= NBS * 2 * HH) return;
  const int c = idx & 127;
  const int n = idx >> 7;
  hcat[idx] = (c < HH) ? ft[(size_t)n * HH + c] : g[(size_t)n * HH + (c - HH)];
}

// out = relu( layernorm(z) * gam + bet ), rows of 64
__global__ void ln_relu_kernel(const float* __restrict__ z,
                               const float* __restrict__ gam,
                               const float* __restrict__ bet,
                               float* __restrict__ out) {
  __shared__ float red[HH];
  const int n = blockIdx.x, j = threadIdx.x;
  const float v = z[(size_t)n * HH + j];
  red[j] = v;
  __syncthreads();
  for (int s = 32; s > 0; s >>= 1) {
    if (j < s) red[j] += red[j + s];
    __syncthreads();
  }
  const float mean = red[0] * (1.0f / HH);
  __syncthreads();
  const float dv = v - mean;
  red[j] = dv * dv;
  __syncthreads();
  for (int s = 32; s > 0; s >>= 1) {
    if (j < s) red[j] += red[j + s];
    __syncthreads();
  }
  const float var = red[0] * (1.0f / HH);
  const float y = dv * rsqrtf(var + 1e-5f) * gam[j] + bet[j];
  out[(size_t)n * HH + j] = fmaxf(y, 0.0f);
}

// logits[n] = sum_j h3[n,j]*W2[j] + b2
__global__ void final_dot_kernel(const float* __restrict__ h3,
                                 const float* __restrict__ W2,
                                 const float* __restrict__ b2,
                                 float* __restrict__ out) {
  __shared__ float red[HH];
  const int n = blockIdx.x, j = threadIdx.x;
  red[j] = h3[(size_t)n * HH + j] * W2[j];
  __syncthreads();
  for (int s = 32; s > 0; s >>= 1) {
    if (j < s) red[j] += red[j + s];
    __syncthreads();
  }
  if (j == 0) out[n] = red[0] + b2[0];
}

// ---------------------------------------------------------------------------
// Host orchestration
// ---------------------------------------------------------------------------
static inline void launch_gemm(const float* A, const float* B, const float* bias,
                               float* C, int M, int K, int N, int act,
                               _Float16* Bpack, hipStream_t stream) {
  packB_kernel<<<(K * N + 255) / 256, 256, 0, stream>>>(B, Bpack, K, N);
  dim3 grid(N / 64, M / 16);
  gemm_wmma4_kernel<<<grid, 32, 0, stream>>>(A, Bpack, bias, C, M, K, N, act);
}

extern "C" void kernel_launch(void* const* d_in, const int* in_sizes, int n_in,
                              void* d_out, int out_size, void* d_ws, size_t ws_size,
                              hipStream_t stream) {
  const float* price        = (const float*)d_in[0];
  const float* tweets       = (const float*)d_in[1];
  const int*   tweet_counts = (const int*)  d_in[2];
  const int*   edge_index   = (const int*)  d_in[3];
  // d_in[4] = num_edge_type (unused by the math)
  const float* pg_Wi = (const float*)d_in[5];
  const float* pg_Wh = (const float*)d_in[6];
  const float* pg_bi = (const float*)d_in[7];
  const float* pg_bh = (const float*)d_in[8];
  const float* tg_Wi = (const float*)d_in[9];
  const float* tg_Wh = (const float*)d_in[10];
  const float* tg_bi = (const float*)d_in[11];
  const float* tg_bh = (const float*)d_in[12];
  const float* dg_Wi = (const float*)d_in[13];
  const float* dg_Wh = (const float*)d_in[14];
  const float* dg_bi = (const float*)d_in[15];
  const float* dg_bh = (const float*)d_in[16];
  const float* bil_A = (const float*)d_in[17];
  const float* bil_b = (const float*)d_in[18];
  const float* gnn_W = (const float*)d_in[19];
  const float* gnn_b = (const float*)d_in[20];
  const float* fc1_W = (const float*)d_in[21];
  const float* fc1_b = (const float*)d_in[22];
  const float* ln1_g = (const float*)d_in[23];
  const float* ln1_b = (const float*)d_in[24];
  const float* fc2_W = (const float*)d_in[25];
  const float* fc2_b = (const float*)d_in[26];
  const float* ln2_g = (const float*)d_in[27];
  const float* ln2_b = (const float*)d_in[28];
  const float* mlp_W1 = (const float*)d_in[29];
  const float* mlp_b1 = (const float*)d_in[30];
  const float* mlp_W2 = (const float*)d_in[31];
  const float* mlp_b2 = (const float*)d_in[32];

  float* ws = (float*)d_ws;
  // BIG region: time-shared by xg_tweet (38.4M) / xg_price (23.04M) + padded
  // price A,B / xg_date (3.84M) / bilinear tmp (16.38M) — stream-ordered.
  float* BIG  = ws;
  size_t off  = (size_t)NTW * KK * G3H;        // 38,400,000 floats
  float* news = ws + off;  off += (size_t)NTW * HH;      // [20000,64]
  float* xprc = ws + off;  off += (size_t)NBS * HH;      // price branch sum
  float* text = ws + off;  off += (size_t)NBS * HH;      // date branch sum
  float* bilBt= ws + off;  off += (size_t)HH * HH * HH;  // [64,4096]
  float* comb = ws + off;  off += (size_t)NBS * HH;      // ft
  float* deg  = ws + off;  off += 512;
  float* agg  = ws + off;  off += (size_t)NBS * HH;
  float* aggn = ws + off;  off += (size_t)NBS * HH;
  float* gbuf = ws + off;  off += (size_t)NBS * HH;
  float* hcat = ws + off;  off += (size_t)NBS * 2 * HH;
  float* zbuf = ws + off;  off += (size_t)NBS * HH;
  float* hbuf = ws + off;  off += (size_t)NBS * HH;
  _Float16* Bpack = (_Float16*)(ws + off);                // 262144 halfs (32B-aligned)
  off += (size_t)HH * HH * HH / 2;                        // 131072 floats

  // Padded price operands live in the tail of BIG (free once tweet GRU is done;
  // xg_price occupies BIG[0 .. 23.04M), pads start at 23.04M).
  float* priceAp = BIG + (size_t)NBS * TT * G3H;          // [120000, 32]
  float* priceBp = priceAp + (size_t)NBS * TT * 32;       // [32, 192]

  float* logits = (float*)d_out;               // [B,S] = 4000 floats

  // ---- Tweet branch: Xg = tweets @ tg_Wi + tg_bi ; ragged GRU -> news ----
  launch_gemm(tweets, tg_Wi, tg_bi, BIG, NTW * KK, TW, G3H, 0, Bpack, stream);
  gru_scan_kernel<<<NTW / 4, 256, 0, stream>>>(BIG, tg_Wh, tg_bh, tweet_counts,
                                               news, NTW, KK);

  // ---- Price branch: pad K 3->32, Xg = priceAp @ priceBp + pg_bi ; GRU ----
  pad_priceA_kernel<<<(NBS * TT * 32 + 255) / 256, 256, 0, stream>>>(price, priceAp);
  pad_priceB_kernel<<<(32 * G3H + 255) / 256, 256, 0, stream>>>(pg_Wi, priceBp);
  launch_gemm(priceAp, priceBp, pg_bi, BIG, NBS * TT, 32, G3H, 0, Bpack, stream);
  gru_scan_kernel<<<NBS / 4, 256, 0, stream>>>(BIG, pg_Wh, pg_bh, nullptr,
                                               xprc, NBS, TT);

  // ---- Date branch: Xg = news @ dg_Wi + dg_bi ; GRU -> text ----
  launch_gemm(news, dg_Wi, dg_bi, BIG, NBS * DD, HH, G3H, 0, Bpack, stream);
  gru_scan_kernel<<<NBS / 4, 256, 0, stream>>>(BIG, dg_Wh, dg_bh, nullptr,
                                               text, NBS, DD);

  // ---- Bilinear fusion: comb = tanh( text A x + b ) ----
  bilA_transpose_kernel<<<(HH * HH * HH + 255) / 256, 256, 0, stream>>>(bil_A, bilBt);
  launch_gemm(text, bilBt, nullptr, BIG, NBS, HH, HH * HH, 0, Bpack, stream);
  bilin_combine_kernel<<<NBS, HH, 0, stream>>>(BIG, xprc, bil_b, comb);

  // ---- GNN: mean-aggregate over edges, then linear+tanh ----
  fill0_kernel<<<(512 + 255) / 256, 256, 0, stream>>>(deg, 512);
  fill0_kernel<<<(NBS * HH + 255) / 256, 256, 0, stream>>>(agg, NBS * HH);
  deg_kernel<<<(EE + 255) / 256, 256, 0, stream>>>(edge_index, deg);
  {
    dim3 grid(EE, BB);
    agg_kernel<<<grid, HH, 0, stream>>>(edge_index, comb, agg);
  }
  agg_norm_kernel<<<NBS, HH, 0, stream>>>(agg, deg, aggn);
  launch_gemm(aggn, gnn_W, gnn_b, gbuf, NBS, HH, HH, 1, Bpack, stream); // tanh

  // ---- Head: concat -> fc1 -> LN+ReLU -> fc2 -> LN+ReLU -> mlp -> logits ----
  concat_kernel<<<(NBS * 2 * HH + 255) / 256, 256, 0, stream>>>(comb, gbuf, hcat);
  launch_gemm(hcat, fc1_W, fc1_b, zbuf, NBS, 2 * HH, HH, 0, Bpack, stream);
  ln_relu_kernel<<<NBS, HH, 0, stream>>>(zbuf, ln1_g, ln1_b, hbuf);
  launch_gemm(hbuf, fc2_W, fc2_b, zbuf, NBS, HH, HH, 0, Bpack, stream);
  ln_relu_kernel<<<NBS, HH, 0, stream>>>(zbuf, ln2_g, ln2_b, hbuf);
  launch_gemm(hbuf, mlp_W1, mlp_b1, zbuf, NBS, HH, HH, 2, Bpack, stream); // relu
  final_dot_kernel<<<NBS, HH, 0, stream>>>(zbuf, mlp_W2, mlp_b2, logits);
}
